// DilatedMHSABlock_53412213293415
// MI455X (gfx1250) — compile-verified, matrix operand order checked by reading, not run on hardware
//
#include <hip/hip_runtime.h>
#include <stdint.h>

// ---------------- problem constants ----------------
#define B_SZ    2
#define N_SEQ   2048
#define D_MODEL 1024
#define N_HEADS 16
#define D_HEAD  64
#define M_ROWS  (B_SZ * N_SEQ)        // 4096
#define K_DIM   D_MODEL               // 1024
#define N_QKV   (3 * D_MODEL)         // 3072

typedef __attribute__((ext_vector_type(16))) __bf16 v16bf;
typedef __attribute__((ext_vector_type(8)))  __bf16 v8bf;
typedef __attribute__((ext_vector_type(8)))  float  v8f;
typedef int v4i_t __attribute__((vector_size(16)));

// ---- CDNA5 async global->LDS path (guarded; falls back to sync copy) ----
#if defined(__has_builtin)
#if __has_builtin(__builtin_amdgcn_global_load_async_to_lds_b128)
#define HAVE_ASYNC_LDS 1
#endif
#if __has_builtin(__builtin_amdgcn_s_wait_asynccnt)
#define HAVE_WAIT_ASYNC 1
#endif
#endif

// builtin signature (from toolchain diagnostic): (v4i AS1*, v4i AS3*, imm, imm)
typedef __attribute__((address_space(1))) v4i_t* gv4i_p;
typedef __attribute__((address_space(3))) v4i_t* lv4i_p;

__device__ inline void async_copy16(const void* g, void* l) {
#if defined(HAVE_ASYNC_LDS)
  __builtin_amdgcn_global_load_async_to_lds_b128((gv4i_p)g, (lv4i_p)l, 0, 0);
#else
  *reinterpret_cast<v8bf*>(l) = *reinterpret_cast<const v8bf*>(g);
#endif
}

__device__ inline void wait_stage() {
#if defined(HAVE_ASYNC_LDS)
#if defined(HAVE_WAIT_ASYNC)
  __builtin_amdgcn_s_wait_asynccnt(0);
#else
  asm volatile("s_wait_asynccnt 0x0" ::: "memory");
#endif
#endif
  __syncthreads();
}

// ---------------- helpers ----------------
__device__ inline __bf16 f2bf(float f) {
  unsigned u = __builtin_bit_cast(unsigned, f);
  unsigned r = (u + 0x7FFFu + ((u >> 16) & 1u)) >> 16;   // round-to-nearest-even
  unsigned short s = (unsigned short)r;
  return __builtin_bit_cast(__bf16, s);
}

// ---------------- kernel 1: f32 -> bf16 convert ----------------
__global__ void cvt_f32_bf16(const float* __restrict__ in, __bf16* __restrict__ out, int n) {
  int i = blockIdx.x * blockDim.x + threadIdx.x;
  if (i < n) out[i] = f2bf(in[i]);
}

// ---------------- GEMM: C = A * W^T + bias (bf16 WMMA, LDS double-buffer) ----
// A: (M x K) bf16 row-major.  W: (N x K) bf16 row-major (B[k][n] = W[n][k]).
// 256 threads = 8 waves (4 in M x 2 in N); block tile 128x128; wave tile 32x64.
#define BM 128
#define BN 128
#define KS 32
#define LDP 40   // LDS row stride in bf16 elements (80 B: 16B-aligned, bank-spread)

// WMMA 16x32 bf16 fragment from an LDS slab (rows x KS, stride LDP):
// lane l holds row (l&15); k-halves 0..7/16..23 (lanes<16), 8..15/24..31 (lanes>=16)
__device__ inline v16bf frag_lds(const __bf16* base, int row0, int lane) {
  const __bf16* p = base + (size_t)(row0 + (lane & 15)) * LDP + ((lane >> 4) << 3);
  v8bf lo = *reinterpret_cast<const v8bf*>(p);
  v8bf hi = *reinterpret_cast<const v8bf*>(p + 16);
  return __builtin_shufflevector(lo, hi, 0,1,2,3,4,5,6,7,8,9,10,11,12,13,14,15);
}

__global__ __launch_bounds__(256)
void gemm_bf16_wmma(const __bf16* __restrict__ A, const __bf16* __restrict__ W,
                    const float* __restrict__ bias, float* __restrict__ C,
                    int Ncols, int K) {
  __shared__ __align__(16) __bf16 sa[2][BM * LDP];
  __shared__ __align__(16) __bf16 sb[2][BN * LDP];

  const int tid  = threadIdx.x;
  const int lane = tid & 31;
  const int w    = tid >> 5;
  const int wm   = (w & 3) * 32;   // wave M offset inside block tile
  const int wn   = (w >> 2) * 64;  // wave N offset inside block tile
  const int m0   = blockIdx.x * BM;
  const int n0   = blockIdx.y * BN;

  v8f acc[2][4];
#pragma unroll
  for (int nt = 0; nt < 4; ++nt) {
    float bv = bias[n0 + wn + nt * 16 + (lane & 15)];
#pragma unroll
    for (int mt = 0; mt < 2; ++mt)
#pragma unroll
      for (int r = 0; r < 8; ++r) acc[mt][nt][r] = bv;
  }

  // stage one 128x32 A slab + 128x32 B slab into LDS buffer `buf`
  auto stage = [&](int buf, int kb) {
#pragma unroll
    for (int it = 0; it < 2; ++it) {
      int c   = tid + it * 256;        // 512 16-byte chunks per operand
      int row = c >> 2;
      int seg = (c & 3) << 3;          // 8 bf16 = 16 B
      async_copy16(A + (size_t)(m0 + row) * K + kb + seg, &sa[buf][row * LDP + seg]);
      async_copy16(W + (size_t)(n0 + row) * K + kb + seg, &sb[buf][row * LDP + seg]);
    }
  };

  stage(0, 0);
  wait_stage();

  const int nsteps = K / KS;
  for (int s = 0; s < nsteps; ++s) {
    const int cur = s & 1;
    if (s + 1 < nsteps) stage(cur ^ 1, (s + 1) * KS);

    v16bf a0 = frag_lds(sa[cur], wm,      lane);
    v16bf a1 = frag_lds(sa[cur], wm + 16, lane);
    v16bf b[4];
#pragma unroll
    for (int nt = 0; nt < 4; ++nt) b[nt] = frag_lds(sb[cur], wn + nt * 16, lane);

#pragma unroll
    for (int nt = 0; nt < 4; ++nt) {
      acc[0][nt] = __builtin_amdgcn_wmma_f32_16x16x32_bf16(
          false, a0, false, b[nt], (short)0, acc[0][nt], false, false);
      acc[1][nt] = __builtin_amdgcn_wmma_f32_16x16x32_bf16(
          false, a1, false, b[nt], (short)0, acc[1][nt], false, false);
    }
    wait_stage();
  }

  // D layout: vgpr r <-> M = r + 8*(lane>>4); N = lane&15
#pragma unroll
  for (int mt = 0; mt < 2; ++mt)
#pragma unroll
    for (int nt = 0; nt < 4; ++nt)
#pragma unroll
      for (int r = 0; r < 8; ++r) {
        int m = m0 + wm + mt * 16 + ((lane >> 4) << 3) + r;
        int n = n0 + wn + nt * 16 + (lane & 15);
        C[(size_t)m * Ncols + n] = acc[mt][nt][r];
      }
}

// ---------------- kernel 3: QK L2-normalize in place ----------------
__global__ void qk_norm(float* __restrict__ qkv) {
  int idx = blockIdx.x * blockDim.x + threadIdx.x;   // over M_ROWS * N_HEADS * 2
  int total = M_ROWS * N_HEADS * 2;
  if (idx >= total) return;
  int which = idx & 1;            // 0 = q, 1 = k
  int h = (idx >> 1) & (N_HEADS - 1);
  int row = idx >> 5;
  float* p = qkv + (size_t)row * N_QKV + which * D_MODEL + h * D_HEAD;
  float s = 0.f;
#pragma unroll
  for (int d = 0; d < D_HEAD; ++d) s += p[d] * p[d];
  float inv = 1.f / (sqrtf(s) + 1e-6f);
#pragma unroll
  for (int d = 0; d < D_HEAD; ++d) p[d] *= inv;
}

// ---------------- kernel 4: dilated windowed attention ----------------
// Block = 64 threads handles (b, h, 64 queries). K/V rows [i0-16, i0+79] in LDS.
#define QT 64
#define KR 96
__global__ __launch_bounds__(64)
void dilated_attn(const float* __restrict__ qkv, __bf16* __restrict__ out) {
  __shared__ float sk[KR][D_HEAD + 1];
  __shared__ float sv[KR][D_HEAD + 1];
  const int t  = threadIdx.x;
  const int i0 = blockIdx.x * QT;
  const int h  = blockIdx.y;
  const int b  = blockIdx.z;
  const float* base = qkv + (size_t)b * N_SEQ * N_QKV;

  for (int jr = 0; jr < KR; ++jr) {
    int j = i0 - 16 + jr;
    bool ok = (j >= 0) && (j < N_SEQ);
    const float* kp = base + (size_t)(ok ? j : 0) * N_QKV + D_MODEL + h * D_HEAD;
    const float* vp = base + (size_t)(ok ? j : 0) * N_QKV + 2 * D_MODEL + h * D_HEAD;
    sk[jr][t] = ok ? kp[t] : 0.f;
    sv[jr][t] = ok ? vp[t] : 0.f;
  }
  __syncthreads();

  const int i = i0 + t;
  float q[D_HEAD];
  const float* qp = base + (size_t)i * N_QKV + h * D_HEAD;
#pragma unroll
  for (int d = 0; d < D_HEAD; ++d) q[d] = qp[d];

  float s[17];
  float mx = -3.0e38f;
#pragma unroll
  for (int jj = 0; jj < 17; ++jj) {
    int j = i + 2 * (jj - 8);
    if (j < 0 || j >= N_SEQ) { s[jj] = -3.0e38f; continue; }
    int jr = t + 2 * jj;                 // j - (i0 - 16)
    float a = 0.f;
#pragma unroll
    for (int d = 0; d < D_HEAD; ++d) a += q[d] * sk[jr][d];
    s[jj] = a;
    mx = fmaxf(mx, a);
  }
  float sum = 0.f;
#pragma unroll
  for (int jj = 0; jj < 17; ++jj) {
    float e = __expf(s[jj] - mx);        // invalid entries underflow to 0
    s[jj] = e;
    sum += e;
  }
  float inv = 1.f / sum;

  __bf16* op = out + ((size_t)(b * N_SEQ + i)) * D_MODEL + h * D_HEAD;
#pragma unroll
  for (int d = 0; d < D_HEAD; ++d) {
    float o = 0.f;
#pragma unroll
    for (int jj = 0; jj < 17; ++jj) o += s[jj] * sv[t + 2 * jj][d];
    op[d] = f2bf(o * inv);
  }
}

// ---------------- host side ----------------
extern "C" void kernel_launch(void* const* d_in, const int* in_sizes, int n_in,
                              void* d_out, int out_size, void* d_ws, size_t ws_size,
                              hipStream_t stream) {
  const float* x      = (const float*)d_in[0];   // (B, N, D)
  const float* qkv_w  = (const float*)d_in[1];   // (3D, D)
  const float* qkv_b  = (const float*)d_in[2];   // (3D,)
  const float* out_w  = (const float*)d_in[3];   // (D, D)
  const float* out_b  = (const float*)d_in[4];   // (D,)
  float* out          = (float*)d_out;           // (B, N, D)

  // workspace layout
  char* ws = (char*)d_ws;
  __bf16* xb    = (__bf16*)ws;  ws += (size_t)M_ROWS * K_DIM * 2;
  __bf16* wqkvb = (__bf16*)ws;  ws += (size_t)N_QKV * K_DIM * 2;
  __bf16* woutb = (__bf16*)ws;  ws += (size_t)D_MODEL * K_DIM * 2;
  float*  qkvf  = (float*)ws;   ws += (size_t)M_ROWS * N_QKV * 4;
  __bf16* attnb = (__bf16*)ws;  ws += (size_t)M_ROWS * D_MODEL * 2;

  // 1. convert inputs to bf16
  {
    int n = M_ROWS * K_DIM;
    cvt_f32_bf16<<<(n + 255) / 256, 256, 0, stream>>>(x, xb, n);
    n = N_QKV * K_DIM;
    cvt_f32_bf16<<<(n + 255) / 256, 256, 0, stream>>>(qkv_w, wqkvb, n);
    n = D_MODEL * K_DIM;
    cvt_f32_bf16<<<(n + 255) / 256, 256, 0, stream>>>(out_w, woutb, n);
  }

  // 2. QKV projection: (4096x1024) @ (3072x1024)^T + bias -> qkvf
  {
    dim3 grid(M_ROWS / BM, N_QKV / BN);
    gemm_bf16_wmma<<<grid, 256, 0, stream>>>(xb, wqkvb, qkv_b, qkvf, N_QKV, K_DIM);
  }

  // 3. QK-norm in place
  {
    int total = M_ROWS * N_HEADS * 2;
    qk_norm<<<(total + 255) / 256, 256, 0, stream>>>(qkvf);
  }

  // 4. dilated windowed attention -> attnb (bf16, (B,N,D) layout)
  {
    dim3 grid(N_SEQ / QT, N_HEADS, B_SZ);
    dilated_attn<<<grid, QT, 0, stream>>>(qkvf, attnb);
  }

  // 5. output projection: (4096x1024) @ (1024x1024)^T + bias -> d_out
  {
    dim3 grid(M_ROWS / BM, D_MODEL / BN);
    gemm_bf16_wmma<<<grid, 256, 0, stream>>>(attnb, woutb, out_b, out, D_MODEL, K_DIM);
  }
}
